// RMSNet_89446988907113
// MI455X (gfx1250) — matrix operand order repaired and checked by reading, not compiled, not run on hardware
//
#include <hip/hip_runtime.h>

// ---------------------------------------------------------------------------
// MI455X (gfx1250) implementation.
// Pipeline: median3x3 -> BN(train) -> fixed bilinear shift -> FC[128->8].
// Shift+BN+FC folded into an effective 8x128 f16 matrix + bias (see analysis).
// Two passes over x (stats, then GEMM) ~ 134 MB @ 23.3 TB/s ~ 6 us.
// GEMM done with v_wmma_f32_16x16x32_f16 (wave32, 16 images per wave).
// ---------------------------------------------------------------------------

typedef __attribute__((ext_vector_type(16))) _Float16 v16h;
typedef __attribute__((ext_vector_type(8)))  float    v8f;

#define NV 8
#define NH 16
#define CH 128          // NV*NH features per image
#define NC 8            // classes
#define BLK 256
#define IMG_STRIDE_U32 68   // 64 used + 4 pad (bank-conflict-free A-frag reads)

union Frag { unsigned u[8]; v16h h; };

__device__ __forceinline__ unsigned pack2h(float a, float b) {
    union { _Float16 h[2]; unsigned u; } r;
    r.h[0] = (_Float16)a; r.h[1] = (_Float16)b;
    return r.u;
}

__device__ __forceinline__ void mnmx(float& a, float& b) {
    float mn = fminf(a, b); b = fmaxf(a, b); a = mn;
}

// Exact median of 9 (rank 4 == torch lower median) via Paeth 19-op network.
__device__ __forceinline__ float median9(float p0, float p1, float p2,
                                         float p3, float p4, float p5,
                                         float p6, float p7, float p8) {
    mnmx(p1, p2); mnmx(p4, p5); mnmx(p7, p8);
    mnmx(p0, p1); mnmx(p3, p4); mnmx(p6, p7);
    mnmx(p1, p2); mnmx(p4, p5); mnmx(p7, p8);
    mnmx(p0, p3); mnmx(p5, p8); mnmx(p4, p7);
    mnmx(p3, p6); mnmx(p1, p4); mnmx(p2, p5);
    mnmx(p4, p7); mnmx(p4, p2); mnmx(p6, p4);
    mnmx(p4, p2);
    return p4;
}

__device__ __forceinline__ void load_row(const float* __restrict__ px, int i, float r[16]) {
    if (i >= 0 && i < NV) {
        const float4* p = (const float4*)(px + i * NH);
        float4 a = p[0], b = p[1], c = p[2], d = p[3];
        r[0]=a.x; r[1]=a.y; r[2]=a.z; r[3]=a.w;
        r[4]=b.x; r[5]=b.y; r[6]=b.z; r[7]=b.w;
        r[8]=c.x; r[9]=c.y; r[10]=c.z; r[11]=c.w;
        r[12]=d.x; r[13]=d.y; r[14]=d.z; r[15]=d.w;
    } else {
        #pragma unroll
        for (int j = 0; j < 16; j++) r[j] = 0.f;
    }
}

// Compute 16 medians for image-row i from the three padded rows.
__device__ __forceinline__ void median_row(const float r0[16], const float r1[16],
                                           const float r2[16], float med[16]) {
    #pragma unroll
    for (int j = 0; j < 16; j++) {
        float a0 = (j > 0)  ? r0[j-1] : 0.f;
        float a2 = (j < 15) ? r0[j+1] : 0.f;
        float b0 = (j > 0)  ? r1[j-1] : 0.f;
        float b2 = (j < 15) ? r1[j+1] : 0.f;
        float c0 = (j > 0)  ? r2[j-1] : 0.f;
        float c2 = (j < 15) ? r2[j+1] : 0.f;
        med[j] = median9(a0, r0[j], a2, b0, r1[j], b2, c0, r2[j], c2);
    }
}

// ---------------------------------------------------------------------------
// K1: per-block partial (sum, sumsq) of median-pooled x. 1 thread = 1 image-row.
// ---------------------------------------------------------------------------
__global__ void median_stats_kernel(const float* __restrict__ x,
                                    float* __restrict__ partial) {
    __shared__ float rs[BLK], rq[BLK];
    const int  tid = threadIdx.x;
    const long gid = (long)blockIdx.x * BLK + tid;
    const long img = gid >> 3;
    const int  i   = (int)(gid & 7);
    const float* px = x + img * CH;

    float r0[16], r1[16], r2[16], med[16];
    load_row(px, i - 1, r0);
    load_row(px, i,     r1);
    load_row(px, i + 1, r2);
    median_row(r0, r1, r2, med);

    float s = 0.f, q = 0.f;
    #pragma unroll
    for (int j = 0; j < 16; j++) { s += med[j]; q += med[j] * med[j]; }

    rs[tid] = s; rq[tid] = q;
    __syncthreads();
    #pragma unroll
    for (int st = BLK / 2; st > 0; st >>= 1) {
        if (tid < st) { rs[tid] += rs[tid + st]; rq[tid] += rq[tid + st]; }
        __syncthreads();
    }
    if (tid == 0) {
        partial[2 * blockIdx.x]     = rs[0];
        partial[2 * blockIdx.x + 1] = rq[0];
    }
}

// ---------------------------------------------------------------------------
// K2 (1 block): reduce partials -> mean/inv_std; fold shift+BN+FC into an
// 8x128 f16 matrix, pre-packed as WMMA B-fragments (32x16, K pairs per VGPR),
// plus folded per-class biases.
// ---------------------------------------------------------------------------
__global__ void prep_kernel(const float* __restrict__ partial, int nblk, float invN,
                            const float* __restrict__ bnw, const float* __restrict__ bnb,
                            const float* __restrict__ xs,  const float* __restrict__ ys,
                            const float* __restrict__ fcw, const float* __restrict__ fcb,
                            unsigned* __restrict__ bfrag, float* __restrict__ biasout) {
    __shared__ float rs[BLK], rq[BLK];
    __shared__ float VV[NC * CH];     // V[n][q] = sum_p fc_w[n,p] * G[p,q]
    __shared__ float stat[2];
    const int tid = threadIdx.x;

    float s = 0.f, q = 0.f;
    for (int i = tid; i < nblk; i += BLK) { s += partial[2*i]; q += partial[2*i + 1]; }
    rs[tid] = s; rq[tid] = q;
    __syncthreads();
    #pragma unroll
    for (int st = BLK / 2; st > 0; st >>= 1) {
        if (tid < st) { rs[tid] += rs[tid + st]; rq[tid] += rq[tid + st]; }
        __syncthreads();
    }
    if (tid == 0) {
        float mean = rs[0] * invN;
        float var  = rq[0] * invN - mean * mean;
        stat[0] = mean;
        stat[1] = rsqrtf(var + 1e-5f);
    }

    // Grid-sample matrix: sample point for output p=(i,j) is
    // (i - yshift - 0.5, j - xshift - 0.5); zero padding outside.
    const float ysv = ys[0], xsv = xs[0];
    for (int r = 0; r < 4; r++) {
        const int e = tid + BLK * r;          // 0..1023 -> (n, q)
        const int n = e >> 7, qq = e & 127;
        const int qi = qq >> 4, qj = qq & 15;
        float acc = 0.f;
        for (int i = 0; i < NV; i++) {
            float iy  = (float)i - ysv - 0.5f;
            float y0f = floorf(iy);
            float wy1 = iy - y0f;
            int   y0  = (int)y0f;
            float wy  = (qi == y0) ? (1.f - wy1) : ((qi == y0 + 1) ? wy1 : 0.f);
            for (int j = 0; j < NH; j++) {
                float ixx = (float)j - xsv - 0.5f;
                float x0f = floorf(ixx);
                float wx1 = ixx - x0f;
                int   x0  = (int)x0f;
                float wx  = (qj == x0) ? (1.f - wx1) : ((qj == x0 + 1) ? wx1 : 0.f);
                acc += wy * wx * fcw[n * CH + i * NH + j];
            }
        }
        VV[e] = acc;
    }
    __syncthreads();

    const float mean = stat[0], inv = stat[1];
    const float sA = inv * bnw[0];            // scale on med
    const float tA = bnb[0] - mean * sA;      // additive affine part

    if (tid < NC) {
        float sv = 0.f;
        for (int k = 0; k < CH; k++) sv += VV[tid * CH + k];
        biasout[tid] = tA * sv + fcb[tid];
    }

    // B-fragment packing used here (32x16 f16, contiguous K, same as A): lane = h*16 + n; VGPR v holds
    // K = h*16 + 2v (lo) and +1 (hi). bfrag[(c*8+v)*32 + lane], c = K-chunk.
    for (int r = 0; r < 4; r++) {
        const int e = tid + BLK * r;          // 0..1023
        const int c = e >> 8, v = (e >> 5) & 7, lane = e & 31;
        const int h = lane >> 4, n = lane & 15;
        float a0 = 0.f, a1 = 0.f;
        if (n < NC) {
            const int k = 32 * c + h * 16 + 2 * v;
            a0 = sA * VV[n * CH + k];
            a1 = sA * VV[n * CH + k + 1];
        }
        bfrag[e] = pack2h(a0, a1);
    }
}

// ---------------------------------------------------------------------------
// K3: median pool (into LDS as f16) + WMMA GEMM. 8 waves/block, 16 images/wave.
// ---------------------------------------------------------------------------
__global__ void __launch_bounds__(BLK)
median_wmma_gemm_kernel(const float* __restrict__ x,
                        const unsigned* __restrict__ bfrag,
                        const float* __restrict__ bias,
                        float* __restrict__ out) {
    __shared__ __align__(16) unsigned medbuf[8 * 16 * IMG_STRIDE_U32]; // 34816 B
    const int  tid  = threadIdx.x;
    const int  wv   = tid >> 5;            // wave id (wave32)
    const int  lane = tid & 31;
    const long imgbase = (long)blockIdx.x * 128 + (long)wv * 16;
    unsigned* mywave = medbuf + wv * 16 * IMG_STRIDE_U32;

    // ---- Phase 1: medians for this wave's 16 images -> LDS (f16 pairs) ----
    #pragma unroll
    for (int k = 0; k < 4; k++) {
        const int tr  = k * 32 + lane;     // tile row 0..127 (contiguous per k)
        const int img = tr >> 3, i = tr & 7;
        const float* px = x + (imgbase + img) * CH;
        float r0[16], r1[16], r2[16], med[16];
        load_row(px, i - 1, r0);
        load_row(px, i,     r1);
        load_row(px, i + 1, r2);
        median_row(r0, r1, r2, med);
        unsigned pk[8];
        #pragma unroll
        for (int jj = 0; jj < 8; jj++) pk[jj] = pack2h(med[2*jj], med[2*jj + 1]);
        uint4* d4 = (uint4*)(mywave + img * IMG_STRIDE_U32 + i * 8);
        d4[0] = make_uint4(pk[0], pk[1], pk[2], pk[3]);
        d4[1] = make_uint4(pk[4], pk[5], pk[6], pk[7]);
    }
    __syncthreads();

    // ---- Phase 2: D[16x16] = A[16x128] * Beff[128x16] via 4 WMMAs ----
    // A order used here (16-bit 16x32, contiguous K, same as B): lane = h*16 + M; VGPR v -> K = 16h+2v.
    const int h   = lane >> 4;
    const int mml = lane & 15;
    const unsigned* arow = mywave + mml * IMG_STRIDE_U32;

    v8f C = {0.f, 0.f, 0.f, 0.f, 0.f, 0.f, 0.f, 0.f};
    #pragma unroll
    for (int c = 0; c < 4; c++) {
        Frag A, B;
        #pragma unroll
        for (int v = 0; v < 8; v++) {
            const int idx = 16 * c + h * 8 + v; // u32 units
            A.u[v] = arow[idx];
            B.u[v] = bfrag[(c * 8 + v) * 32 + lane];
        }
        C = __builtin_amdgcn_wmma_f32_16x16x32_f16(
                /*neg_a=*/false, A.h, /*neg_b=*/false, B.h,
                /*c_mod=*/(short)0, C, /*reuse_a=*/false, /*reuse_b=*/false);
    }

    // C/D layout: VGPR r, lanes 0-15 -> (M=r, N=lane); lanes 16-31 -> (M=8+r).
    if (mml < NC) {
        const float bs = bias[mml];
        const long rowbase = imgbase + h * 8;
        #pragma unroll
        for (int r = 0; r < 8; r++)
            out[(rowbase + r) * NC + mml] = C[r] + bs;
    }
}

// ---------------------------------------------------------------------------
extern "C" void kernel_launch(void* const* d_in, const int* in_sizes, int n_in,
                              void* d_out, int out_size, void* d_ws, size_t ws_size,
                              hipStream_t stream) {
    const float* x   = (const float*)d_in[0];   // [B,1,8,16]
    const float* bnw = (const float*)d_in[1];
    const float* bnb = (const float*)d_in[2];
    const float* xs  = (const float*)d_in[3];
    const float* ys  = (const float*)d_in[4];
    const float* fcw = (const float*)d_in[5];   // [8,128]
    const float* fcb = (const float*)d_in[6];   // [8]
    float* out = (float*)d_out;

    const long nimg  = (long)in_sizes[0] / CH;  // 131072
    const int  nblk1 = (int)(nimg * NV / BLK);  // 4096 stats blocks
    const int  nblk3 = (int)(nimg / 128);       // 1024 gemm blocks

    // ws layout: [2*nblk1 floats partials][1024 u32 bfrag][8 floats bias]
    float*    partial = (float*)d_ws;
    size_t    off     = ((size_t)2 * nblk1 * sizeof(float) + 255) & ~(size_t)255;
    unsigned* bfrag   = (unsigned*)((char*)d_ws + off);
    float*    biasb   = (float*)((char*)d_ws + off + 1024 * sizeof(unsigned));
    const float invN  = 1.0f / ((float)nimg * (float)CH);

    median_stats_kernel<<<nblk1, BLK, 0, stream>>>(x, partial);
    prep_kernel<<<1, BLK, 0, stream>>>(partial, nblk1, invN, bnw, bnb, xs, ys,
                                       fcw, fcb, bfrag, biasb);
    median_wmma_gemm_kernel<<<nblk3, BLK, 0, stream>>>(x, bfrag, biasb, out);
}
